// MMDKernel_40785009442873
// MI455X (gfx1250) — compile-verified, matrix operand order checked by reading, not run on hardware
//
#include <hip/hip_runtime.h>

typedef __attribute__((ext_vector_type(16))) __bf16 v16bf;
typedef __attribute__((ext_vector_type(8)))  __bf16 v8bf;
typedef __attribute__((ext_vector_type(8)))  float  v8f;

#define NN 8192
#define DD 256
#define T16 (NN / 16)            // 512  (16-wide tiles, fallback path)
#define T32 (NN / 32)            // 256  (32-wide tiles, main path)
#define KCH (DD / 32)            // 8 k-chunks of 32
#define NUM_BW 5

// ---------------- workspace layout ----------------
// [0, 32)              : double acc[4]  (3 pairing accumulators)
// [64, 64+65536)       : float snorm[8192], tnorm[8192]
// [BF_OFF, ...)        : __bf16 Shi, Slo, Thi, Tlo  (each NN*DD elements)
#define NORM_OFF 64
#define BF_OFF (64 + (size_t)(2 * NN) * 4)

// -------- fragment loads (ISA 16-bit A-matrix 16x32 layout) --------
// lane L<16 holds row M=L, K = kb..kb+7 and kb+16..kb+23 (kb = chunk base)
// lane L>=16 holds row M=L-16, K = kb+8..kb+15 and kb+24..kb+31
// handled by caller via ksel = (lane>>4)*8 added to kb.

__device__ inline v16bf load_frag_pre(const __bf16* __restrict__ base, int row, int kb) {
    const __bf16* p = base + (size_t)row * DD + kb;
    v8bf a = *(const v8bf*)(p);
    v8bf b = *(const v8bf*)(p + 16);
    return __builtin_shufflevector(a, b, 0,1,2,3,4,5,6,7,8,9,10,11,12,13,14,15);
}

__device__ inline void load_frag_f32(const float* __restrict__ base, int row, int kb,
                                     v16bf& hi, v16bf& lo) {
    const float* p = base + (size_t)row * DD + kb;
    float f[16];
    *(float4*)(f + 0)  = *(const float4*)(p + 0);
    *(float4*)(f + 4)  = *(const float4*)(p + 4);
    *(float4*)(f + 8)  = *(const float4*)(p + 16);
    *(float4*)(f + 12) = *(const float4*)(p + 20);
#pragma unroll
    for (int i = 0; i < 16; ++i) {
        __bf16 h = (__bf16)f[i];
        hi[i] = h;
        lo[i] = (__bf16)(f[i] - (float)h);
    }
}

// ---------------- helper kernels ----------------

__global__ void init_kernel(double* __restrict__ acc) {
    if (threadIdx.x < 4) acc[threadIdx.x] = 0.0;
}

__global__ void cvt_kernel(const float* __restrict__ in, __bf16* __restrict__ hi,
                           __bf16* __restrict__ lo, int n) {
    int i = blockIdx.x * blockDim.x + threadIdx.x;
    if (i < n) {
        float f = in[i];
        __bf16 h = (__bf16)f;
        hi[i] = h;
        lo[i] = (__bf16)(f - (float)h);
    }
}

__global__ void norm_kernel(const float* __restrict__ S, const float* __restrict__ T,
                            float* __restrict__ snorm, float* __restrict__ tnorm) {
    int i = blockIdx.x * blockDim.x + threadIdx.x;   // 0 .. 2*NN-1
    const float* src = (i < NN) ? (S + (size_t)i * DD) : (T + (size_t)(i - NN) * DD);
    float acc = 0.f;
#pragma unroll 4
    for (int k = 0; k < DD; k += 4) {
        float4 v = *(const float4*)(src + k);
        acc += v.x * v.x + v.y * v.y + v.z * v.z + v.w * v.w;
    }
    if (i < NN) snorm[i] = acc; else tnorm[i - NN] = acc;
}

// ----- shared epilogue: one 16x16 accumulator -> partial sum of 5 exp kernels -----
__device__ inline float tile_epilogue(const v8f& c, const float* __restrict__ xn,
                                      float yv, int m0, const float* coef) {
    float s = 0.f;
#pragma unroll
    for (int e = 0; e < 8; ++e) {
        float d = xn[m0 + e] + yv - 2.0f * c[e];
        d = fmaxf(d, 0.0f);
#pragma unroll
        for (int j = 0; j < NUM_BW; ++j)
            s += __expf(-d * coef[j]);
    }
    return s;
}

__device__ inline void select_pair(int pairing,
                                   const float* S, const float* T,
                                   const __bf16* Shi, const __bf16* Slo,
                                   const __bf16* Thi, const __bf16* Tlo,
                                   const float* snorm, const float* tnorm,
                                   const float*& X, const float*& Y,
                                   const __bf16*& Xhi, const __bf16*& Xlo,
                                   const __bf16*& Yhi, const __bf16*& Ylo,
                                   const float*& xn, const float*& yn) {
    if (pairing == 0)      { X=S; Y=S; Xhi=Shi; Xlo=Slo; Yhi=Shi; Ylo=Slo; xn=snorm; yn=snorm; }
    else if (pairing == 1) { X=T; Y=T; Xhi=Thi; Xlo=Tlo; Yhi=Thi; Ylo=Tlo; xn=tnorm; yn=tnorm; }
    else                   { X=S; Y=T; Xhi=Shi; Xlo=Slo; Yhi=Thi; Ylo=Tlo; xn=snorm; yn=tnorm; }
}

// ---------------- main kernel (bf16-split precomputed): 32x32 tile per wave ----------------
// grid: (T32*T32/8, 3), block = 256 threads = 8 waves arranged 2x4 over 64x128 region.
// K-loop unrolled only 2x: double-buffers loads against WMMAs while keeping
// peak VGPR liveness ~170 (no spills, no VGPR-MSB addressing).

__global__ __launch_bounds__(256) void mmd_main_pre(
    const float* __restrict__ bw,
    const __bf16* __restrict__ Shi, const __bf16* __restrict__ Slo,
    const __bf16* __restrict__ Thi, const __bf16* __restrict__ Tlo,
    const float* __restrict__ snorm, const float* __restrict__ tnorm,
    double* __restrict__ acc)
{
    const int pairing = blockIdx.y;          // 0=SS 1=TT 2=ST
    const int wave = threadIdx.x >> 5;
    const int lane = threadIdx.x & 31;

    // 2x4 wave arrangement of 32x32 tiles -> block covers 64 rows x 128 cols
    const int bi = blockIdx.x / (T32 / 4);   // 0 .. T32/2-1
    const int bj = blockIdx.x % (T32 / 4);   // 0 .. T32/4-1
    const int ti = bi * 2 + (wave >> 2);
    const int tj = bj * 4 + (wave & 3);

    const float *X, *Y, *xn, *yn;
    const __bf16 *Xhi, *Xlo, *Yhi, *Ylo;
    select_pair(pairing, nullptr, nullptr, Shi, Slo, Thi, Tlo, snorm, tnorm,
                X, Y, Xhi, Xlo, Yhi, Ylo, xn, yn);

    const int ksel  = (lane >> 4) * 8;
    const int arow0 = ti * 32 + (lane & 15);
    const int arow1 = arow0 + 16;
    const int brow0 = tj * 32 + (lane & 15);
    const int brow1 = brow0 + 16;

    v8f c00 = {0.f,0.f,0.f,0.f,0.f,0.f,0.f,0.f};
    v8f c01 = c00, c10 = c00, c11 = c00;

#pragma unroll 2
    for (int kc = 0; kc < KCH; ++kc) {
        const int kb = kc * 32 + ksel;
        v16bf ah0 = load_frag_pre(Xhi, arow0, kb);
        v16bf al0 = load_frag_pre(Xlo, arow0, kb);
        v16bf ah1 = load_frag_pre(Xhi, arow1, kb);
        v16bf al1 = load_frag_pre(Xlo, arow1, kb);
        v16bf bh0 = load_frag_pre(Yhi, brow0, kb);
        v16bf bl0 = load_frag_pre(Ylo, brow0, kb);
        v16bf bh1 = load_frag_pre(Yhi, brow1, kb);
        v16bf bl1 = load_frag_pre(Ylo, brow1, kb);
        // bf16x3 compensated fp32 products for 4 sub-tiles
        c00 = __builtin_amdgcn_wmma_f32_16x16x32_bf16(false, ah0, false, bh0, (short)0, c00, false, false);
        c00 = __builtin_amdgcn_wmma_f32_16x16x32_bf16(false, ah0, false, bl0, (short)0, c00, false, false);
        c00 = __builtin_amdgcn_wmma_f32_16x16x32_bf16(false, al0, false, bh0, (short)0, c00, false, false);
        c01 = __builtin_amdgcn_wmma_f32_16x16x32_bf16(false, ah0, false, bh1, (short)0, c01, false, false);
        c01 = __builtin_amdgcn_wmma_f32_16x16x32_bf16(false, ah0, false, bl1, (short)0, c01, false, false);
        c01 = __builtin_amdgcn_wmma_f32_16x16x32_bf16(false, al0, false, bh1, (short)0, c01, false, false);
        c10 = __builtin_amdgcn_wmma_f32_16x16x32_bf16(false, ah1, false, bh0, (short)0, c10, false, false);
        c10 = __builtin_amdgcn_wmma_f32_16x16x32_bf16(false, ah1, false, bl0, (short)0, c10, false, false);
        c10 = __builtin_amdgcn_wmma_f32_16x16x32_bf16(false, al1, false, bh0, (short)0, c10, false, false);
        c11 = __builtin_amdgcn_wmma_f32_16x16x32_bf16(false, ah1, false, bh1, (short)0, c11, false, false);
        c11 = __builtin_amdgcn_wmma_f32_16x16x32_bf16(false, ah1, false, bl1, (short)0, c11, false, false);
        c11 = __builtin_amdgcn_wmma_f32_16x16x32_bf16(false, al1, false, bh1, (short)0, c11, false, false);
    }

    float coef[NUM_BW];
#pragma unroll
    for (int j = 0; j < NUM_BW; ++j) {
        float b = bw[j];
        coef[j] = 1.0f / (2.0f * b * b);
    }

    // C/D layout: lanes 0-15 -> N=lane, M=vgpr; lanes 16-31 -> N=lane-16, M=8+vgpr
    const float yv0 = yn[tj * 32 + (lane & 15)];
    const float yv1 = yn[tj * 32 + 16 + (lane & 15)];
    const int   m0  = ti * 32 + ((lane >> 4) << 3);

    float s = 0.f;
    s += tile_epilogue(c00, xn, yv0, m0,      coef);
    s += tile_epilogue(c01, xn, yv1, m0,      coef);
    s += tile_epilogue(c10, xn, yv0, m0 + 16, coef);
    s += tile_epilogue(c11, xn, yv1, m0 + 16, coef);

#pragma unroll
    for (int off = 16; off > 0; off >>= 1)
        s += __shfl_xor(s, off, 32);
    if (lane == 0)
        atomicAdd(&acc[pairing], (double)s);
}

// ---------------- fallback kernel (inline fp32->bf16 split): 16x16 per wave ----------------

__global__ __launch_bounds__(256) void mmd_main_raw(
    const float* __restrict__ S, const float* __restrict__ T,
    const float* __restrict__ bw,
    const float* __restrict__ snorm, const float* __restrict__ tnorm,
    double* __restrict__ acc)
{
    const int pairing = blockIdx.y;
    const int wave = threadIdx.x >> 5;
    const int lane = threadIdx.x & 31;
    const int tile = blockIdx.x * 8 + wave;
    const int ti = tile / T16;
    const int tj = tile % T16;

    const float *X, *Y, *xn, *yn;
    const __bf16 *d0, *d1, *d2, *d3;
    select_pair(pairing, S, T, nullptr, nullptr, nullptr, nullptr, snorm, tnorm,
                X, Y, d0, d1, d2, d3, xn, yn);

    const int arow = ti * 16 + (lane & 15);
    const int brow = tj * 16 + (lane & 15);
    const int ksel = (lane >> 4) * 8;

    v8f c = {0.f,0.f,0.f,0.f,0.f,0.f,0.f,0.f};
#pragma unroll 2
    for (int kc = 0; kc < KCH; ++kc) {
        const int kb = kc * 32 + ksel;
        v16bf ahi, alo, bhi, blo;
        load_frag_f32(X, arow, kb, ahi, alo);
        load_frag_f32(Y, brow, kb, bhi, blo);
        c = __builtin_amdgcn_wmma_f32_16x16x32_bf16(false, ahi, false, bhi, (short)0, c, false, false);
        c = __builtin_amdgcn_wmma_f32_16x16x32_bf16(false, ahi, false, blo, (short)0, c, false, false);
        c = __builtin_amdgcn_wmma_f32_16x16x32_bf16(false, alo, false, bhi, (short)0, c, false, false);
    }

    float coef[NUM_BW];
#pragma unroll
    for (int j = 0; j < NUM_BW; ++j) {
        float b = bw[j];
        coef[j] = 1.0f / (2.0f * b * b);
    }
    const float yv = yn[tj * 16 + (lane & 15)];
    const int   m0 = ti * 16 + ((lane >> 4) << 3);
    float s = tile_epilogue(c, xn, yv, m0, coef);

#pragma unroll
    for (int off = 16; off > 0; off >>= 1)
        s += __shfl_xor(s, off, 32);
    if (lane == 0)
        atomicAdd(&acc[pairing], (double)s);
}

__global__ void fin_kernel(const double* __restrict__ acc, float* __restrict__ out) {
    if (threadIdx.x == 0) {
        const double nm = (double)NN * (double)NN * (double)NUM_BW;
        double kss = acc[0] / nm;
        double ktt = acc[1] / nm;
        double kst = acc[2] / nm;
        out[0] = (float)(kss + ktt - 2.0 * kst);
    }
}

// ---------------- launcher ----------------

extern "C" void kernel_launch(void* const* d_in, const int* in_sizes, int n_in,
                              void* d_out, int out_size, void* d_ws, size_t ws_size,
                              hipStream_t stream) {
    const float* S  = (const float*)d_in[0];
    const float* T  = (const float*)d_in[1];
    const float* bw = (const float*)d_in[2];
    float* out = (float*)d_out;

    char* ws = (char*)d_ws;
    double* acc  = (double*)ws;
    float* snorm = (float*)(ws + NORM_OFF);
    float* tnorm = snorm + NN;

    __bf16* Shi = (__bf16*)(ws + BF_OFF);
    __bf16* Slo = Shi + (size_t)NN * DD;
    __bf16* Thi = Slo + (size_t)NN * DD;
    __bf16* Tlo = Thi + (size_t)NN * DD;
    const size_t need = BF_OFF + 4ull * NN * DD * sizeof(__bf16);
    const bool pre = ws_size >= need;

    init_kernel<<<1, 64, 0, stream>>>(acc);
    norm_kernel<<<(2 * NN) / 256, 256, 0, stream>>>(S, T, snorm, tnorm);

    if (pre) {
        const int n = NN * DD;
        cvt_kernel<<<(n + 255) / 256, 256, 0, stream>>>(S, Shi, Slo, n);
        cvt_kernel<<<(n + 255) / 256, 256, 0, stream>>>(T, Thi, Tlo, n);
        dim3 grid((T32 / 2) * (T32 / 4), 3);   // 8 waves = 2x4 tiles of 32x32
        mmd_main_pre<<<grid, 256, 0, stream>>>(bw, Shi, Slo, Thi, Tlo, snorm, tnorm, acc);
    } else {
        dim3 grid(T16 * T16 / 8, 3);
        mmd_main_raw<<<grid, 256, 0, stream>>>(S, T, bw, snorm, tnorm, acc);
    }

    fin_kernel<<<1, 32, 0, stream>>>(acc, out);
}
